// MultiHeadAttention_57586921504919
// MI455X (gfx1250) — matrix operand order repaired and checked by reading, not compiled, not run on hardware
//
#include <hip/hip_runtime.h>
#include <hip/hip_bf16.h>

#define TB 4
#define TT 2048
#define TE 1024
#define TH 16
#define TD 64
#define THD 1024  // TH*TD

typedef __bf16 bf16;
typedef __attribute__((ext_vector_type(16))) __bf16 v16bf;
typedef __attribute__((ext_vector_type(8)))  __bf16 v8bf;
typedef __attribute__((ext_vector_type(8)))  float   v8f;

static __device__ __forceinline__ v8f wmma_bf16(v16bf a, v16bf b, v8f c) {
  // (neg_a, A, neg_b, B, c_mod, C, reuse_a, reuse_b)
  return __builtin_amdgcn_wmma_f32_16x16x32_bf16(false, a, false, b, (short)0, c, false, false);
}

// A-fragment (16xK row-major source): lane L -> row m0+(L%16);
// elem e<8 -> K = k0 + 8*(L/16) + e ; elem e>=8 -> K = k0 + 16 + 8*(L/16) + (e-8)
static __device__ __forceinline__ v16bf load_a_frag(const bf16* A, int ld, int m0, int k0, int lane) {
  const bf16* p = A + (size_t)(m0 + (lane & 15)) * ld + k0 + ((lane >> 4) << 3);
  v8bf lo = *(const v8bf*)p;
  v8bf hi = *(const v8bf*)(p + 16);
  v16bf r;
#pragma unroll
  for (int i = 0; i < 8; ++i) { r[i] = lo[i]; r[i + 8] = hi[i]; }
  return r;
}

// B-fragment from Bt stored [N][K] row-major: lane L -> col n0+(L%16);
// elem e<8 -> K = k0 + 8*(L/16) + e ; elem e>=8 -> K = k0 + 16 + 8*(L/16) + (e-8)
static __device__ __forceinline__ v16bf load_b_frag(const bf16* Bt, int ld, int n0, int k0, int lane) {
  const bf16* p = Bt + (size_t)(n0 + (lane & 15)) * ld + k0 + ((lane >> 4) << 3);
  v8bf lo = *(const v8bf*)p;
  v8bf hi = *(const v8bf*)(p + 16);
  v16bf r;
#pragma unroll
  for (int i = 0; i < 8; ++i) { r[i] = lo[i]; r[i + 8] = hi[i]; }
  return r;
}

__global__ void cvt_bf16_kernel(const float* __restrict__ src, bf16* __restrict__ dst, int n) {
  for (int i = blockIdx.x * blockDim.x + threadIdx.x; i < n; i += gridDim.x * blockDim.x)
    dst[i] = (bf16)src[i];
}

// src is [rowsK][colsN] fp32 ; dst is [colsN][rowsK] bf16 (i.e. [N][K])
__global__ void transpose_bf16_kernel(const float* __restrict__ src, bf16* __restrict__ dst,
                                      int rowsK, int colsN) {
  int n = rowsK * colsN;
  for (int i = blockIdx.x * blockDim.x + threadIdx.x; i < n; i += gridDim.x * blockDim.x) {
    int k = i / colsN;
    int c = i - k * colsN;
    dst[(size_t)c * rowsK + k] = (bf16)src[i];
  }
}

// Fused Q/K/V projection. One wave = one 32x64 tile of one of the 3 outputs.
// Q,K stored [b][h][t][d]; V stored transposed [b][h][d][t].
__global__ __launch_bounds__(256) void qkv_proj_kernel(
    const bf16* __restrict__ xb,
    const bf16* __restrict__ wqt, const bf16* __restrict__ wkt, const bf16* __restrict__ wvt,
    bf16* __restrict__ Q, bf16* __restrict__ K, bf16* __restrict__ Vt) {
  int lane = threadIdx.x & 31;
  int wave = (blockIdx.x * blockDim.x + threadIdx.x) >> 5;
  int mat = wave >> 12;        // 0=Q 1=K 2=V (4096 tiles each)
  int rem = wave & 4095;
  int mt = rem >> 4;           // 256 row tiles of 32
  int h  = rem & 15;           // head == 64-wide n tile
  int m0 = mt << 5;
  int n0 = h << 6;
  const bf16* Bt = (mat == 0) ? wqt : (mat == 1) ? wkt : wvt;

  v8f c00 = {}, c01 = {}, c02 = {}, c03 = {};
  v8f c10 = {}, c11 = {}, c12 = {}, c13 = {};
  for (int k0 = 0; k0 < TE; k0 += 32) {
    __builtin_prefetch(xb + (size_t)(m0 + (lane & 15)) * TE + k0 + 64, 0, 1);
    __builtin_prefetch(Bt + (size_t)(n0 + (lane & 15)) * TE + k0 + 64, 0, 1);
    v16bf a0 = load_a_frag(xb, TE, m0,      k0, lane);
    v16bf a1 = load_a_frag(xb, TE, m0 + 16, k0, lane);
    v16bf b0 = load_b_frag(Bt, TE, n0 +  0, k0, lane);
    v16bf b1 = load_b_frag(Bt, TE, n0 + 16, k0, lane);
    v16bf b2 = load_b_frag(Bt, TE, n0 + 32, k0, lane);
    v16bf b3 = load_b_frag(Bt, TE, n0 + 48, k0, lane);
    c00 = wmma_bf16(a0, b0, c00);  c10 = wmma_bf16(a1, b0, c10);
    c01 = wmma_bf16(a0, b1, c01);  c11 = wmma_bf16(a1, b1, c11);
    c02 = wmma_bf16(a0, b2, c02);  c12 = wmma_bf16(a1, b2, c12);
    c03 = wmma_bf16(a0, b3, c03);  c13 = wmma_bf16(a1, b3, c13);
  }

  int colA = lane & 15, rowOff = (lane >> 4) << 3;
  v8f cc[2][4] = {{c00, c01, c02, c03}, {c10, c11, c12, c13}};
#pragma unroll
  for (int g = 0; g < 2; ++g) {
#pragma unroll
    for (int f = 0; f < 4; ++f) {
      int d = (f << 4) + colA;
#pragma unroll
      for (int r = 0; r < 8; ++r) {
        int m = m0 + (g << 4) + rowOff + r;
        int t = m & (TT - 1);
        int bb = m >> 11;
        bf16 v = (bf16)cc[g][f][r];
        if (mat == 2)
          Vt[(((size_t)bb * TH + h) * TD + d) * TT + t] = v;
        else {
          bf16* dst = (mat == 0) ? Q : K;
          dst[(((size_t)bb * TH + h) * TT + t) * TD + d] = v;
        }
      }
    }
  }
}

// Flash attention: one wave per (b,h,16-query tile). 32 keys per step.
__global__ __launch_bounds__(256) void attn_kernel(
    const bf16* __restrict__ Qb, const bf16* __restrict__ Kb,
    const bf16* __restrict__ Vt, bf16* __restrict__ ctx) {
  __shared__ bf16 pshared[8 * 16 * 32];   // 1KB per wave of P (16x32 bf16)
  int lane = threadIdx.x & 31;
  int warp = threadIdx.x >> 5;
  bf16* pb = pshared + warp * (16 * 32);

  int wave = blockIdx.x * 8 + warp;
  int qt = wave & 127;
  int h  = (wave >> 7) & 15;
  int bb = wave >> 11;
  int q0 = qt << 4;

  const bf16* Qbh = Qb + ((size_t)bb * TH + h) * TT * TD;
  const bf16* Kbh = Kb + ((size_t)bb * TH + h) * TT * TD;
  const bf16* Vbh = Vt + ((size_t)bb * TH + h) * TD * TT;  // [d][t]

  v16bf aq0 = load_a_frag(Qbh, TD, q0, 0, lane);
  v16bf aq1 = load_a_frag(Qbh, TD, q0, 32, lane);

  float mrow[8], lrow[8];
  v8f o0 = {}, o1 = {}, o2 = {}, o3 = {};
#pragma unroll
  for (int r = 0; r < 8; ++r) { mrow[r] = -1e30f; lrow[r] = 0.f; }

  int colA = lane & 15, rowOff = (lane >> 4) << 3;
  const float scale = 0.125f;   // 1/sqrt(64)
  int kend = q0 + 16;           // causal: keys 0..q0+15

  for (int kt = 0; kt < kend; kt += 32) {
    v8f s0 = {}, s1 = {};
    s0 = wmma_bf16(aq0, load_b_frag(Kbh, TD, kt,       0, lane), s0);
    s0 = wmma_bf16(aq1, load_b_frag(Kbh, TD, kt,      32, lane), s0);
    s1 = wmma_bf16(aq0, load_b_frag(Kbh, TD, kt + 16,  0, lane), s1);
    s1 = wmma_bf16(aq1, load_b_frag(Kbh, TD, kt + 16, 32, lane), s1);

#pragma unroll
    for (int r = 0; r < 8; ++r) {
      int q = q0 + rowOff + r;
      float v0 = (kt + colA      <= q) ? s0[r] * scale : -1e30f;
      float v1 = (kt + 16 + colA <= q) ? s1[r] * scale : -1e30f;
      float mx = fmaxf(v0, v1);
      mx = fmaxf(mx, __shfl_xor(mx, 1, 16));
      mx = fmaxf(mx, __shfl_xor(mx, 2, 16));
      mx = fmaxf(mx, __shfl_xor(mx, 4, 16));
      mx = fmaxf(mx, __shfl_xor(mx, 8, 16));
      float mnew = fmaxf(mrow[r], mx);
      float p0 = __expf(v0 - mnew);
      float p1 = __expf(v1 - mnew);
      float alpha = __expf(mrow[r] - mnew);
      float ps = p0 + p1;
      ps += __shfl_xor(ps, 1, 16);
      ps += __shfl_xor(ps, 2, 16);
      ps += __shfl_xor(ps, 4, 16);
      ps += __shfl_xor(ps, 8, 16);
      lrow[r] = lrow[r] * alpha + ps;
      mrow[r] = mnew;
      o0[r] *= alpha; o1[r] *= alpha; o2[r] *= alpha; o3[r] *= alpha;
      pb[(rowOff + r) * 32 + colA]      = (bf16)p0;
      pb[(rowOff + r) * 32 + 16 + colA] = (bf16)p1;
    }

    asm volatile("s_wait_dscnt 0" ::: "memory");

    // Re-read P as an A-fragment (16x32)
    const bf16* pr = pb + (size_t)(lane & 15) * 32 + ((lane >> 4) << 3);
    v8bf plo = *(const v8bf*)pr;
    v8bf phi = *(const v8bf*)(pr + 16);
    v16bf pa;
#pragma unroll
    for (int i = 0; i < 8; ++i) { pa[i] = plo[i]; pa[i + 8] = phi[i]; }

    o0 = wmma_bf16(pa, load_b_frag(Vbh, TT,  0, kt, lane), o0);
    o1 = wmma_bf16(pa, load_b_frag(Vbh, TT, 16, kt, lane), o1);
    o2 = wmma_bf16(pa, load_b_frag(Vbh, TT, 32, kt, lane), o2);
    o3 = wmma_bf16(pa, load_b_frag(Vbh, TT, 48, kt, lane), o3);
  }

  v8f oo[4] = {o0, o1, o2, o3};
#pragma unroll
  for (int r = 0; r < 8; ++r) {
    float inv = 1.0f / lrow[r];
    int q = q0 + rowOff + r;
    size_t base = ((size_t)bb * TT + q) * THD + (size_t)h * TD;
#pragma unroll
    for (int f = 0; f < 4; ++f)
      ctx[base + (f << 4) + colA] = (bf16)(oo[f][r] * inv);
  }
}

// out = ctx @ Wo + bo  (fp32 out). One wave per 32x64 tile.
__global__ __launch_bounds__(256) void out_proj_kernel(
    const bf16* __restrict__ ctxb, const bf16* __restrict__ wot,
    const float* __restrict__ bo, float* __restrict__ out) {
  int lane = threadIdx.x & 31;
  int wave = (blockIdx.x * blockDim.x + threadIdx.x) >> 5;
  int mt = wave >> 4;          // 256 row tiles of 32
  int nt = wave & 15;
  int m0 = mt << 5, n0 = nt << 6;

  v8f c00 = {}, c01 = {}, c02 = {}, c03 = {};
  v8f c10 = {}, c11 = {}, c12 = {}, c13 = {};
  for (int k0 = 0; k0 < THD; k0 += 32) {
    __builtin_prefetch(ctxb + (size_t)(m0 + (lane & 15)) * THD + k0 + 64, 0, 1);
    v16bf a0 = load_a_frag(ctxb, THD, m0,      k0, lane);
    v16bf a1 = load_a_frag(ctxb, THD, m0 + 16, k0, lane);
    v16bf b0 = load_b_frag(wot, THD, n0 +  0, k0, lane);
    v16bf b1 = load_b_frag(wot, THD, n0 + 16, k0, lane);
    v16bf b2 = load_b_frag(wot, THD, n0 + 32, k0, lane);
    v16bf b3 = load_b_frag(wot, THD, n0 + 48, k0, lane);
    c00 = wmma_bf16(a0, b0, c00);  c10 = wmma_bf16(a1, b0, c10);
    c01 = wmma_bf16(a0, b1, c01);  c11 = wmma_bf16(a1, b1, c11);
    c02 = wmma_bf16(a0, b2, c02);  c12 = wmma_bf16(a1, b2, c12);
    c03 = wmma_bf16(a0, b3, c03);  c13 = wmma_bf16(a1, b3, c13);
  }

  int colA = lane & 15, rowOff = (lane >> 4) << 3;
  v8f cc[2][4] = {{c00, c01, c02, c03}, {c10, c11, c12, c13}};
#pragma unroll
  for (int g = 0; g < 2; ++g) {
#pragma unroll
    for (int f = 0; f < 4; ++f) {
      int n = n0 + (f << 4) + colA;
      float bias = bo[n];
#pragma unroll
      for (int r = 0; r < 8; ++r) {
        int m = m0 + (g << 4) + rowOff + r;
        out[(size_t)m * TE + n] = cc[g][f][r] + bias;
      }
    }
  }
}

extern "C" void kernel_launch(void* const* d_in, const int* in_sizes, int n_in,
                              void* d_out, int out_size, void* d_ws, size_t ws_size,
                              hipStream_t stream) {
  const float* x  = (const float*)d_in[0];
  const float* Wq = (const float*)d_in[1];
  const float* Wk = (const float*)d_in[2];
  const float* Wv = (const float*)d_in[3];
  const float* Wo = (const float*)d_in[4];
  const float* bo = (const float*)d_in[5];
  float* out = (float*)d_out;

  char* ws = (char*)d_ws;
  const size_t XB_B  = (size_t)TB * TT * TE * 2;       // 16MB
  const size_t W_B   = (size_t)TE * THD * 2;           // 2MB each
  const size_t QKV_B = (size_t)TB * TH * TT * TD * 2;  // 16MB each
  bf16* xb  = (bf16*)(ws);
  bf16* wqt = (bf16*)(ws + XB_B);
  bf16* wkt = (bf16*)(ws + XB_B + W_B);
  bf16* wvt = (bf16*)(ws + XB_B + 2 * W_B);
  bf16* wot = (bf16*)(ws + XB_B + 3 * W_B);
  bf16* Q   = (bf16*)(ws + XB_B + 4 * W_B);
  bf16* K   = (bf16*)(ws + XB_B + 4 * W_B + QKV_B);
  bf16* Vt  = (bf16*)(ws + XB_B + 4 * W_B + 2 * QKV_B);
  bf16* ctx = (bf16*)(ws + XB_B + 4 * W_B + 3 * QKV_B);

  cvt_bf16_kernel<<<4096, 256, 0, stream>>>(x, xb, TB * TT * TE);
  transpose_bf16_kernel<<<2048, 256, 0, stream>>>(Wq, wqt, TE, THD);
  transpose_bf16_kernel<<<2048, 256, 0, stream>>>(Wk, wkt, TE, THD);
  transpose_bf16_kernel<<<2048, 256, 0, stream>>>(Wv, wvt, TE, THD);
  transpose_bf16_kernel<<<2048, 256, 0, stream>>>(Wo, wot, THD, TE);

  // 3 * 256 * 16 tiles / 8 waves per block
  qkv_proj_kernel<<<1536, 256, 0, stream>>>(xb, wqt, wkt, wvt, Q, K, Vt);
  // 4 * 16 * 128 q-tiles / 8 waves per block
  attn_kernel<<<1024, 256, 0, stream>>>(Q, K, Vt, ctx);
  // 256 * 16 tiles / 8 waves per block
  out_proj_kernel<<<512, 256, 0, stream>>>(ctx, wot, bo, out);
}
